// DifferentialAttention_80582176408186
// MI455X (gfx1250) — matrix operand order repaired and checked by reading, not compiled, not run on hardware
//
#include <hip/hip_runtime.h>
#include <hip/hip_bf16.h>
#include <math.h>

// Problem constants (from reference)
#define B_    4
#define N_    2048
#define C_    512
#define H_    8
#define DH_   64
#define SCALE_ 0.125f   // DH^-0.5

typedef __attribute__((ext_vector_type(16))) __bf16 v16bf;
typedef __attribute__((ext_vector_type(8)))  float  v8f;

union Frag {
    v16bf v;
    unsigned short u[16];
    unsigned int   w[8];
};

// Hardware convert (v_cvt_bf16_f32), not a bit trick.
__device__ __forceinline__ unsigned short f2bf(float f) {
    __bf16 h = (__bf16)f;
    return __builtin_bit_cast(unsigned short, h);
}
__device__ __forceinline__ unsigned int pack2bf(float a, float b) {
    return (unsigned int)f2bf(a) | ((unsigned int)f2bf(b) << 16);
}

// 16-bit A-matrix (16x32) element->K mapping for a lane half (contiguous: k = 16*half + e, matching B).
__device__ __forceinline__ int a_kidx(int e, int half) {
    return (half ? 16 : 0) + e;
}

__device__ __forceinline__ v8f wmma_bf16(const Frag& a, const Frag& b, v8f c) {
    return __builtin_amdgcn_wmma_f32_16x16x32_bf16(false, a.v, false, b.v,
                                                   (short)0, c, false, false);
}

__device__ __forceinline__ float rsum16(float v) {
    v += __shfl_xor(v, 8, 16);
    v += __shfl_xor(v, 4, 16);
    v += __shfl_xor(v, 2, 16);
    v += __shfl_xor(v, 1, 16);
    return v;
}

// Merge 16 per-lane partial softmax states (M = running max, Z = running sum
// scaled by e^-M) into the full-row state, replicated across the 16 lanes.
__device__ __forceinline__ void merge16(float& M, float& Z) {
#pragma unroll
    for (int d = 8; d >= 1; d >>= 1) {
        const float Mo = __shfl_xor(M, d, 16);
        const float Zo = __shfl_xor(Z, d, 16);
        const float nm = fmaxf(M, Mo);
        Z = Z * __expf(M - nm) + Zo * __expf(Mo - nm);
        M = nm;
    }
}

// ---------------------------------------------------------------------------
// Kernel 0: one-shot fp32 -> bf16 conversion (memory bound, hardware cvt).
// ---------------------------------------------------------------------------
__global__ void __launch_bounds__(256)
cvt_bf16_kernel(const float* __restrict__ src, unsigned short* __restrict__ dst,
                int n)
{
    const int i = (blockIdx.x * 256 + threadIdx.x) * 4;
    if (i < n) {
        float4 f = *(const float4*)(src + i);
        uint2 o;
        o.x = pack2bf(f.x, f.y);
        o.y = pack2bf(f.z, f.w);
        *(uint2*)(dst + i) = o;
    }
}

// ---------------------------------------------------------------------------
// Kernel 1: QKV projection. One wave computes a 16x64 tile: one A-fragment is
// reused across 4 WMMAs per k-step. 64 output columns == exactly one
// (chunk, head): chunks 0..4 = q1,k1,v,q2,k2. q pre-scaled by DH^-0.5.
// V is stored transposed: VT[B,H,DH,N] so P@V B-frags load contiguously.
// ---------------------------------------------------------------------------
__global__ void __launch_bounds__(32)
qkv_gemm_kernel(const unsigned short* __restrict__ xb,   // [8192,512] bf16
                const unsigned short* __restrict__ wb,   // [2560,512] bf16
                unsigned short* __restrict__ q1, unsigned short* __restrict__ k1,
                unsigned short* __restrict__ vt, unsigned short* __restrict__ q2,
                unsigned short* __restrict__ k2)
{
    const int lane = threadIdx.x;
    const int half = lane >> 4, lq = lane & 15;
    const int jt = blockIdx.x;           // 0..39 (groups of 64 output cols)
    const int mt = blockIdx.y;           // 0..511

    const unsigned short* arow = xb + (size_t)(mt * 16 + lq) * C_;

    v8f acc[4] = {v8f{}, v8f{}, v8f{}, v8f{}};
    for (int k0 = 0; k0 < C_; k0 += 32) {
        Frag a;
#pragma unroll
        for (int p = 0; p < 8; ++p)
            a.w[p] = *(const unsigned int*)(arow + k0 + a_kidx(2 * p, half));
#pragma unroll
        for (int sub = 0; sub < 4; ++sub) {
            const unsigned short* wr =
                wb + (size_t)(jt * 64 + sub * 16 + lq) * C_ + k0 + half * 16;
            Frag b;
#pragma unroll
            for (int p = 0; p < 8; ++p)
                b.w[p] = *(const unsigned int*)(wr + 2 * p);
            acc[sub] = wmma_bf16(a, b, acc[sub]);
        }
    }

    const int chunk = (jt * 64) >> 9;           // uniform per block
    const int h     = ((jt * 64) & 511) >> 6;   // uniform per block
    unsigned short* dst;
    float scl = 1.0f;
    bool isV = false;
    switch (chunk) {
        case 0:  dst = q1; scl = SCALE_; break;
        case 1:  dst = k1; break;
        case 2:  dst = vt; isV = true; break;
        case 3:  dst = q2; scl = SCALE_; break;
        default: dst = k2; break;
    }
#pragma unroll
    for (int sub = 0; sub < 4; ++sub) {
#pragma unroll
        for (int r = 0; r < 8; ++r) {
            const int d  = sub * 16 + lq;
            const int m  = mt * 16 + r + 8 * half;
            const int bb = m >> 11, n = m & (N_ - 1);
            const size_t off = isV
                ? ((((size_t)bb * H_ + h) * DH_ + d) * N_ + n)
                : ((((size_t)bb * H_ + h) * N_ + n) * DH_ + d);
            dst[off] = f2bf(acc[sub][r] * scl);
        }
    }
}

// ---------------------------------------------------------------------------
// Kernel 2: fused differential attention, one wave per (b,h,16-row tile).
// Pass 1: PER-LANE online (max,sum) for s1,s2 — no cross-lane ops in the
// loop; the 16 partial states per row are merged once at the end.
// Pass 2: a_i = exp(s_i - M_i)/Z_i; t = (1+l)a1 - l*a2 is bounded in
// [-l, 1+l] so the second softmax needs no max pass: p = exp(t), normalize
// at the end. P tile round-trips LDS to convert C/D layout -> A layout.
// ---------------------------------------------------------------------------
__global__ void __launch_bounds__(32)
diff_attn_kernel(const unsigned short* __restrict__ q1g,
                 const unsigned short* __restrict__ k1g,
                 const unsigned short* __restrict__ vtg,   // [B,H,DH,N]
                 const unsigned short* __restrict__ q2g,
                 const unsigned short* __restrict__ k2g,
                 const float* __restrict__ lambda_p,
                 unsigned short* __restrict__ og)          // [B,N,C] bf16
{
    __shared__ unsigned short ldsP[16 * 32];

    const int lane = threadIdx.x;
    const int half = lane >> 4, lq = lane & 15;
    const int nt = blockIdx.x;
    const int h  = blockIdx.y;
    const int b  = blockIdx.z;
    const int n0 = nt * 16;
    const size_t bh = ((size_t)b * H_ + h) * (size_t)N_ * DH_;

    const unsigned short* Q1 = q1g + bh;
    const unsigned short* K1 = k1g + bh;
    const unsigned short* VT = vtg + bh;   // [DH][N]
    const unsigned short* Q2 = q2g + bh;
    const unsigned short* K2 = k2g + bh;

    const float lam = lambda_p[0];

    // Q A-fragments (row = n0+lq, K = dh as two 32-chunks), loaded once.
    Frag qa1[2], qa2[2];
#pragma unroll
    for (int c = 0; c < 2; ++c) {
        const unsigned short* r1 = Q1 + (size_t)(n0 + lq) * DH_ + c * 32;
        const unsigned short* r2 = Q2 + (size_t)(n0 + lq) * DH_ + c * 32;
#pragma unroll
        for (int p = 0; p < 8; ++p) {
            const int k = a_kidx(2 * p, half);
            qa1[c].w[p] = *(const unsigned int*)(r1 + k);
            qa2[c].w[p] = *(const unsigned int*)(r2 + k);
        }
    }

    float M1[8], Z1[8], M2[8], Z2[8];
#pragma unroll
    for (int r = 0; r < 8; ++r) { M1[r] = -1e30f; Z1[r] = 0.f; M2[r] = -1e30f; Z2[r] = 0.f; }

    // ---- Pass 1: per-lane online softmax statistics (no cross-lane ops) ----
    for (int m0 = 0; m0 < N_; m0 += 32) {
        v8f s1a = {}, s1b = {}, s2a = {}, s2b = {};
#pragma unroll
        for (int t = 0; t < 2; ++t) {
            const int mc = m0 + t * 16;
            Frag b1[2], b2[2];
#pragma unroll
            for (int c = 0; c < 2; ++c) {
                const unsigned short* r1 = K1 + (size_t)(mc + lq) * DH_ + c * 32 + half * 16;
                const unsigned short* r2 = K2 + (size_t)(mc + lq) * DH_ + c * 32 + half * 16;
#pragma unroll
                for (int p = 0; p < 8; ++p) {
                    b1[c].w[p] = *(const unsigned int*)(r1 + 2 * p);
                    b2[c].w[p] = *(const unsigned int*)(r2 + 2 * p);
                }
            }
            if (t == 0) {
                s1a = wmma_bf16(qa1[0], b1[0], s1a);
                s1a = wmma_bf16(qa1[1], b1[1], s1a);
                s2a = wmma_bf16(qa2[0], b2[0], s2a);
                s2a = wmma_bf16(qa2[1], b2[1], s2a);
            } else {
                s1b = wmma_bf16(qa1[0], b1[0], s1b);
                s1b = wmma_bf16(qa1[1], b1[1], s1b);
                s2b = wmma_bf16(qa2[0], b2[0], s2b);
                s2b = wmma_bf16(qa2[1], b2[1], s2b);
            }
        }
#pragma unroll
        for (int r = 0; r < 8; ++r) {
            {
                const float v0 = s1a[r], v1 = s1b[r];
                const float nm = fmaxf(M1[r], fmaxf(v0, v1));
                Z1[r] = Z1[r] * __expf(M1[r] - nm)
                      + __expf(v0 - nm) + __expf(v1 - nm);
                M1[r] = nm;
            }
            {
                const float v0 = s2a[r], v1 = s2b[r];
                const float nm = fmaxf(M2[r], fmaxf(v0, v1));
                Z2[r] = Z2[r] * __expf(M2[r] - nm)
                      + __expf(v0 - nm) + __expf(v1 - nm);
                M2[r] = nm;
            }
        }
    }

    // Single cross-lane merge of the 16 per-lane partial states per row.
#pragma unroll
    for (int r = 0; r < 8; ++r) {
        merge16(M1[r], Z1[r]);
        merge16(M2[r], Z2[r]);
    }

    float iZ1[8], iZ2[8], Zp[8];
#pragma unroll
    for (int r = 0; r < 8; ++r) { iZ1[r] = 1.0f / Z1[r]; iZ2[r] = 1.0f / Z2[r]; Zp[r] = 0.f; }

    v8f o[4] = {v8f{}, v8f{}, v8f{}, v8f{}};   // 4 dh groups of 16
    const float c1 = 1.0f + lam, c2 = -lam;

    // ---- Pass 2: recompute scores, p = exp((1+l)a1 - l*a2), P @ V ----
    for (int m0 = 0; m0 < N_; m0 += 32) {
#pragma unroll
        for (int t = 0; t < 2; ++t) {
            const int mc = m0 + t * 16;
            Frag b1[2], b2[2];
#pragma unroll
            for (int c = 0; c < 2; ++c) {
                const unsigned short* r1 = K1 + (size_t)(mc + lq) * DH_ + c * 32 + half * 16;
                const unsigned short* r2 = K2 + (size_t)(mc + lq) * DH_ + c * 32 + half * 16;
#pragma unroll
                for (int p = 0; p < 8; ++p) {
                    b1[c].w[p] = *(const unsigned int*)(r1 + 2 * p);
                    b2[c].w[p] = *(const unsigned int*)(r2 + 2 * p);
                }
            }
            v8f s1 = {}, s2 = {};
            s1 = wmma_bf16(qa1[0], b1[0], s1);
            s1 = wmma_bf16(qa1[1], b1[1], s1);
            s2 = wmma_bf16(qa2[0], b2[0], s2);
            s2 = wmma_bf16(qa2[1], b2[1], s2);
#pragma unroll
            for (int r = 0; r < 8; ++r) {
                const float a1v = __expf(s1[r] - M1[r]) * iZ1[r];
                const float a2v = __expf(s2[r] - M2[r]) * iZ2[r];
                const float pv  = __expf(c1 * a1v + c2 * a2v);  // bounded, no max
                Zp[r] += pv;
                ldsP[(r + 8 * half) * 32 + t * 16 + lq] = f2bf(pv);
            }
        }
        __syncthreads();

        // A fragment of P (16x32) from LDS.
        Frag pa;
#pragma unroll
        for (int p = 0; p < 8; ++p)
            pa.w[p] = *(const unsigned int*)&ldsP[lq * 32 + a_kidx(2 * p, half)];

        // B fragments of V from VT[DH][N]: col dh fixed per lane, K = m
        // contiguous -> dword loads.
#pragma unroll
        for (int g = 0; g < 4; ++g) {
            const unsigned short* vr =
                VT + (size_t)(g * 16 + lq) * N_ + m0 + half * 16;
            Frag vb;
#pragma unroll
            for (int p = 0; p < 8; ++p)
                vb.w[p] = *(const unsigned int*)(vr + 2 * p);
            o[g] = wmma_bf16(pa, vb, o[g]);
        }
        __syncthreads();
    }

    float invZ[8];
#pragma unroll
    for (int r = 0; r < 8; ++r) invZ[r] = 1.0f / rsum16(Zp[r]);
#pragma unroll
    for (int g = 0; g < 4; ++g) {
#pragma unroll
        for (int r = 0; r < 8; ++r) {
            const int n = n0 + r + 8 * half;
            og[((size_t)b * N_ + n) * C_ + h * DH_ + g * 16 + lq] =
                f2bf(o[g][r] * invZ[r]);
        }
    }
}

// ---------------------------------------------------------------------------
// Kernel 3: output projection out = O @ proj_w^T + proj_b (fp32 out).
// One wave computes a 16x64 tile (A-frag reused across 4 WMMAs per k-step).
// ---------------------------------------------------------------------------
__global__ void __launch_bounds__(32)
proj_gemm_kernel(const unsigned short* __restrict__ og,   // [8192,512] bf16
                 const unsigned short* __restrict__ pwb,  // [512,512] bf16
                 const float* __restrict__ bias, float* __restrict__ out)
{
    const int lane = threadIdx.x;
    const int half = lane >> 4, lq = lane & 15;
    const int jt = blockIdx.x;            // 0..7 (groups of 64 cols)
    const int mt = blockIdx.y;            // 0..511

    const unsigned short* arow = og + (size_t)(mt * 16 + lq) * C_;

    v8f acc[4] = {v8f{}, v8f{}, v8f{}, v8f{}};
    for (int k0 = 0; k0 < C_; k0 += 32) {
        Frag a;
#pragma unroll
        for (int p = 0; p < 8; ++p)
            a.w[p] = *(const unsigned int*)(arow + k0 + a_kidx(2 * p, half));
#pragma unroll
        for (int sub = 0; sub < 4; ++sub) {
            const unsigned short* wr =
                pwb + (size_t)(jt * 64 + sub * 16 + lq) * C_ + k0 + half * 16;
            Frag b;
#pragma unroll
            for (int p = 0; p < 8; ++p)
                b.w[p] = *(const unsigned int*)(wr + 2 * p);
            acc[sub] = wmma_bf16(a, b, acc[sub]);
        }
    }

#pragma unroll
    for (int sub = 0; sub < 4; ++sub) {
        const int j = jt * 64 + sub * 16 + lq;
        const float bj = bias[j];
#pragma unroll
        for (int r = 0; r < 8; ++r) {
            const int m = mt * 16 + r + 8 * half;
            out[(size_t)m * C_ + j] = acc[sub][r] + bj;
        }
    }
}

// ---------------------------------------------------------------------------
extern "C" void kernel_launch(void* const* d_in, const int* in_sizes, int n_in,
                              void* d_out, int out_size, void* d_ws, size_t ws_size,
                              hipStream_t stream)
{
    (void)in_sizes; (void)n_in; (void)out_size; (void)ws_size;
    const float* x      = (const float*)d_in[0];
    const float* qkv_w  = (const float*)d_in[1];
    const float* proj_w = (const float*)d_in[2];
    const float* proj_b = (const float*)d_in[3];
    const float* lam    = (const float*)d_in[4];
    float* out = (float*)d_out;

    // Workspace layout (bf16 elements):
    const size_t nX  = (size_t)B_ * N_ * C_;          // xb
    const size_t nW  = (size_t)2560 * C_;             // wb (used rows)
    const size_t nPW = (size_t)C_ * C_;               // pwb
    const size_t per = (size_t)B_ * H_ * N_ * DH_;    // each of q1..k2
    unsigned short* xb  = (unsigned short*)d_ws;
    unsigned short* wb  = xb + nX;
    unsigned short* pwb = wb + nW;
    unsigned short* q1  = pwb + nPW;
    unsigned short* k1  = q1 + per;
    unsigned short* vt  = k1 + per;                   // transposed [B,H,DH,N]
    unsigned short* q2  = vt + per;
    unsigned short* k2  = q2 + per;
    unsigned short* og  = k2 + per;                   // [B*N, C] bf16

    const dim3 blk32(32, 1, 1);

    // 0) fp32 -> bf16 conversions (once).
    {
        const int tx = 256, vec = 4;
        hipLaunchKernelGGL(cvt_bf16_kernel, dim3((nX / vec + tx - 1) / tx), dim3(tx),
                           0, stream, x, xb, (int)nX);
        hipLaunchKernelGGL(cvt_bf16_kernel, dim3((nW / vec + tx - 1) / tx), dim3(tx),
                           0, stream, qkv_w, wb, (int)nW);
        hipLaunchKernelGGL(cvt_bf16_kernel, dim3((nPW / vec + tx - 1) / tx), dim3(tx),
                           0, stream, proj_w, pwb, (int)nPW);
    }

    // 1) QKV projection over the 2560 used output columns (40 groups of 64).
    dim3 g1(2560 / 64, (B_ * N_) / 16, 1);
    hipLaunchKernelGGL(qkv_gemm_kernel, g1, blk32, 0, stream,
                       xb, wb, q1, k1, vt, q2, k2);

    // 2) Fused differential attention.
    dim3 g2(N_ / 16, H_, B_);
    hipLaunchKernelGGL(diff_attn_kernel, g2, blk32, 0, stream,
                       q1, k1, vt, q2, k2, lam, og);

    // 3) Output projection + bias.
    dim3 g3(C_ / 64, (B_ * N_) / 16, 1);
    hipLaunchKernelGGL(proj_gemm_kernel, g3, blk32, 0, stream,
                       og, pwb, proj_b, out);
}
